// Transformer_24953759990334
// MI455X (gfx1250) — compile-verified, matrix operand order checked by reading, not run on hardware
//
#include <hip/hip_runtime.h>
#include <hip/hip_bf16.h>
#include <hip/hip_fp16.h>
#include <math.h>

typedef __attribute__((ext_vector_type(16))) _Float16 v16h;
typedef __attribute__((ext_vector_type(8)))  float    v8f;

constexpr int DIM    = 128;
constexpr int NHEAD  = 8;
constexpr int NB     = 4;
constexpr int TLEN   = 2048;
constexpr int WID    = 512;
constexpr int NOUT   = 32000;
constexpr int NCHUNK = 32;              // split-K chunks over timesteps
constexpr int SLEN   = TLEN / NCHUNK;   // 64 timesteps per chunk
constexpr int TS     = 32;              // tile rows staged in LDS
constexpr int HSS    = 132;             // LDS row stride (floats): 528B, 16B-aligned

// ---------------------------------------------------------------------------
// K1: h_last = wte[x[b, T-1]] ; Q[b,h,e] = sum_d h_last[d] * A[h,d,e]
// grid = NB*NHEAD blocks, 128 threads
// ---------------------------------------------------------------------------
__global__ void k_embed_q(const int* __restrict__ x, const float* __restrict__ A,
                          const float* __restrict__ wte, float* __restrict__ Q) {
    const int b    = blockIdx.x >> 3;
    const int head = blockIdx.x & 7;
    const int tid  = threadIdx.x;           // 0..127
    __shared__ float hl[DIM];
    const int idx = x[b * TLEN + (TLEN - 1)];
    hl[tid] = wte[idx * DIM + tid];
    __syncthreads();
    const float* Ah = A + head * DIM * DIM;
    float s = 0.f;
    for (int d = 0; d < DIM; ++d) s += hl[d] * Ah[d * DIM + tid];
    Q[(b * NHEAD + head) * DIM + tid] = s;
}

// ---------------------------------------------------------------------------
// K2a: split-K online-softmax attention partials for the last token.
// grid = NB*NCHUNK blocks, 256 threads = 8 waves; wave w handles head w.
// wte rows gathered tile-by-tile into LDS via global_load_async_to_lds_b128
// (one 16B segment per lane, no VGPR round-trip), ASYNCcnt-waited + barrier.
// Emits partial (m, l, acc[128]) per (b, head, chunk).
// ---------------------------------------------------------------------------
__global__ void k_attn_part(const int* __restrict__ x, const float* __restrict__ wte,
                            const float* __restrict__ Q,
                            float* __restrict__ macc,   // [b][h][ch][128]
                            float* __restrict__ ml) {   // [b][h][ch][2]
    const int b      = blockIdx.x / NCHUNK;
    const int ch     = blockIdx.x % NCHUNK;
    const int s_base = ch * SLEN;
    const int tid    = threadIdx.x;         // 0..255
    __shared__ float QL[NHEAD * DIM];
    __shared__ float Hs[TS * HSS];
    for (int i = tid; i < NHEAD * DIM; i += 256) QL[i] = Q[b * NHEAD * DIM + i];

    const int head = tid >> 5;
    const int lane = tid & 31;
    float m = -INFINITY, l = 0.f;
    float a0 = 0.f, a1 = 0.f, a2 = 0.f, a3 = 0.f;

    for (int t = 0; t < SLEN; t += TS) {
        __syncthreads();
        // Async gather: TS rows x 32 x 16B segments = 1024 segs, 4 per thread.
        for (int i = tid; i < TS * 32; i += 256) {
            const int row = i >> 5, seg = i & 31;
            const int idx = x[b * TLEN + s_base + t + row];
            const float* g = wte + (size_t)idx * DIM + seg * 4;
            const unsigned loff = (unsigned)(size_t)&Hs[row * HSS + seg * 4];
            asm volatile("global_load_async_to_lds_b128 %0, %1, off"
                         :: "v"(loff), "v"(g) : "memory");
        }
        asm volatile("s_wait_asynccnt 0x0" ::: "memory");
        __syncthreads();

        // score for (head, s = s_base + t + lane)
        const float* q  = &QL[head * DIM];
        const float* hr = &Hs[lane * HSS];
        float sc = 0.f;
        for (int k = 0; k < DIM; ++k) sc += q[k] * hr[k];

        float smax = sc;
        for (int off = 16; off; off >>= 1) smax = fmaxf(smax, __shfl_xor(smax, off));
        const float m_new = fmaxf(m, smax);
        const float scale = __expf(m - m_new);
        const float p     = __expf(sc - m_new);
        float psum = p;
        for (int off = 16; off; off >>= 1) psum += __shfl_xor(psum, off);
        l = l * scale + psum;
        a0 *= scale; a1 *= scale; a2 *= scale; a3 *= scale;
        for (int j = 0; j < TS; ++j) {
            const float pj = __shfl(p, j);
            const float* hj = &Hs[j * HSS];
            a0 += pj * hj[lane];
            a1 += pj * hj[lane + 32];
            a2 += pj * hj[lane + 64];
            a3 += pj * hj[lane + 96];
        }
        m = m_new;
    }
    const int pbase = ((b * NHEAD + head) * NCHUNK + ch);
    float* pa = macc + (size_t)pbase * DIM;
    pa[lane]      = a0;
    pa[lane + 32] = a1;
    pa[lane + 64] = a2;
    pa[lane + 96] = a3;
    if (lane == 0) { ml[pbase * 2] = m; ml[pbase * 2 + 1] = l; }
}

// ---------------------------------------------------------------------------
// K2b: combine split-K partials: M=max m_i; o = sum w_i*acc_i / sum w_i*l_i.
// grid = NB*NHEAD blocks, 128 threads.
// ---------------------------------------------------------------------------
__global__ void k_attn_combine(const float* __restrict__ macc,
                               const float* __restrict__ ml,
                               float* __restrict__ O) {
    const int b    = blockIdx.x >> 3;
    const int head = blockIdx.x & 7;
    const int tid  = threadIdx.x;           // 0..127
    __shared__ float sm[NCHUNK], sl[NCHUNK];
    const int pb = (b * NHEAD + head) * NCHUNK;
    if (tid < NCHUNK) { sm[tid] = ml[(pb + tid) * 2]; sl[tid] = ml[(pb + tid) * 2 + 1]; }
    __syncthreads();
    float M = -INFINITY;
    for (int c = 0; c < NCHUNK; ++c) M = fmaxf(M, sm[c]);
    float L = 0.f, acc = 0.f;
    for (int c = 0; c < NCHUNK; ++c) {
        const float w = __expf(sm[c] - M);
        L   += w * sl[c];
        acc += w * macc[(size_t)(pb + c) * DIM + tid];
    }
    O[(b * NHEAD + head) * DIM + tid] = acc / L;
}

// ---------------------------------------------------------------------------
// K3: xl = o @ V ; z = relu(xl@W1 + b1) @ W2 ; xl += z
// Writes xl as a 16x128 f16 tile (rows 4..15 zeroed) for the WMMA GEMM.
// grid = NB blocks, 128 threads.
// ---------------------------------------------------------------------------
__global__ void k_mlp(const float* __restrict__ O, const float* __restrict__ V,
                      const float* __restrict__ W1, const float* __restrict__ b1,
                      const float* __restrict__ W2, _Float16* __restrict__ xl16) {
    const int b   = blockIdx.x;
    const int tid = threadIdx.x;            // 0..127
    __shared__ float oL[NHEAD * DIM];
    __shared__ float xlL[DIM];
    __shared__ float hidL[WID];
    for (int i = tid; i < NHEAD * DIM; i += DIM) oL[i] = O[b * NHEAD * DIM + i];
    __syncthreads();
    float s = 0.f;
    for (int k = 0; k < NHEAD * DIM; ++k) s += oL[k] * V[k * DIM + tid];
    xlL[tid] = s;
    __syncthreads();
    for (int w = tid; w < WID; w += DIM) {
        float hv = b1[w];
        for (int d = 0; d < DIM; ++d) hv += xlL[d] * W1[d * WID + w];
        hidL[w] = fmaxf(hv, 0.f);
    }
    __syncthreads();
    float z = 0.f;
    for (int w = 0; w < WID; ++w) z += hidL[w] * W2[w * DIM + tid];
    const float xf = z + xlL[tid];
    xl16[b * DIM + tid]        = (_Float16)xf;
    xl16[(b + 4) * DIM + tid]  = (_Float16)0.f;   // zero padding rows 4..15
    xl16[(b + 8) * DIM + tid]  = (_Float16)0.f;
    xl16[(b + 12) * DIM + tid] = (_Float16)0.f;
}

// ---------------------------------------------------------------------------
// K4: logits = XL(16x128) @ unembed^T(128x16 tile) via v_wmma_f32_16x16x32_f16.
// One wave per 16-column tile; 4 WMMAs over K=128.
// grid = NOUT/16 = 2000 blocks, 32 threads.
// ---------------------------------------------------------------------------
__global__ void k_logits(const _Float16* __restrict__ xl16,
                         const float* __restrict__ unembed,
                         float* __restrict__ logits) {
    const int col0 = blockIdx.x * 16;
    const int lane = threadIdx.x;           // 0..31
    const int mn   = lane & 15;             // A row index / B col index
    const int hi   = lane >> 4;             // lane-group K split
    const _Float16* arow = xl16 + mn * DIM;
    const float*    ucol = unembed + (size_t)(col0 + mn) * DIM;  // B[k][n] = U[n][k]

    v8f c = {};
    for (int kc = 0; kc < DIM; kc += 32) {
        v16h a, bm;
#pragma unroll
        for (int e = 0; e < 16; ++e) {
            const int K = (e < 8) ? (kc + hi * 8 + e)
                                  : (kc + 16 + hi * 8 + (e - 8));
            a[e]  = arow[K];
            bm[e] = (_Float16)ucol[K];
        }
        c = __builtin_amdgcn_wmma_f32_16x16x32_f16(
                /*neg_a=*/false, a, /*neg_b=*/false, bm,
                /*c_mod=*/(short)0, c, /*reuse_a=*/false, /*reuse_b=*/false);
    }
    if (hi == 0) {                          // rows 0..3 valid (M = r, N = lane)
#pragma unroll
        for (int r = 0; r < 4; ++r)
            logits[(size_t)r * NOUT + col0 + lane] = c[r];
    }
}

// ---------------------------------------------------------------------------
// K5: row softmax over 32000 -> d_out. grid = NB blocks, 512 threads.
// ---------------------------------------------------------------------------
__global__ void k_softmax(const float* __restrict__ logits, float* __restrict__ out) {
    const int b   = blockIdx.x;
    const int tid = threadIdx.x;            // 0..511
    __shared__ float red[512];
    const float* row = logits + (size_t)b * NOUT;
    float m = -INFINITY;
    for (int j = tid; j < NOUT; j += 512) m = fmaxf(m, row[j]);
    red[tid] = m; __syncthreads();
    for (int s = 256; s; s >>= 1) { if (tid < s) red[tid] = fmaxf(red[tid], red[tid + s]); __syncthreads(); }
    m = red[0]; __syncthreads();
    float sum = 0.f;
    for (int j = tid; j < NOUT; j += 512) sum += __expf(row[j] - m);
    red[tid] = sum; __syncthreads();
    for (int s = 256; s; s >>= 1) { if (tid < s) red[tid] += red[tid + s]; __syncthreads(); }
    const float inv = 1.f / red[0];
    for (int j = tid; j < NOUT; j += 512) out[(size_t)b * NOUT + j] = __expf(row[j] - m) * inv;
}

// ---------------------------------------------------------------------------
// ws layout (floats):
//   Q      [0,      4096)
//   O      [4096,   8192)
//   xl16   [8192,   9216)    (2048 halfs)
//   logits [9216,   137216)
//   macc   [137216, 268288)  (4*8*32*128)
//   ml     [268288, 270336)  (4*8*32*2)
// total ~1.06 MB
// ---------------------------------------------------------------------------
extern "C" void kernel_launch(void* const* d_in, const int* in_sizes, int n_in,
                              void* d_out, int out_size, void* d_ws, size_t ws_size,
                              hipStream_t stream) {
    const int*   x       = (const int*)  d_in[0];
    const float* A       = (const float*)d_in[1];
    const float* V       = (const float*)d_in[2];
    const float* wte     = (const float*)d_in[3];
    const float* unembed = (const float*)d_in[4];
    const float* W1      = (const float*)d_in[5];
    const float* b1      = (const float*)d_in[6];
    const float* W2      = (const float*)d_in[7];
    float* outp = (float*)d_out;

    float*    ws     = (float*)d_ws;
    float*    Q      = ws;                      // 4096 floats
    float*    O      = ws + 4096;               // 4096 floats
    _Float16* xl16   = (_Float16*)(ws + 8192);  // 2048 halfs
    float*    logits = ws + 9216;               // 128000 floats
    float*    macc   = ws + 137216;             // 131072 floats
    float*    ml     = ws + 268288;             // 2048 floats

    k_embed_q    <<<dim3(NB * NHEAD),  dim3(DIM), 0, stream>>>(x, A, wte, Q);
    k_attn_part  <<<dim3(NB * NCHUNK), dim3(256), 0, stream>>>(x, wte, Q, macc, ml);
    k_attn_combine<<<dim3(NB * NHEAD), dim3(DIM), 0, stream>>>(macc, ml, O);
    k_mlp        <<<dim3(NB),          dim3(DIM), 0, stream>>>(O, V, W1, b1, W2, xl16);
    k_logits     <<<dim3(NOUT / 16),   dim3(32),  0, stream>>>(xl16, unembed, logits);
    k_softmax    <<<dim3(NB),          dim3(512), 0, stream>>>(logits, outp);
}
